// LSTM_8615704395881
// MI455X (gfx1250) — compile-verified, matrix operand order checked by reading, not run on hardware
//
#include <hip/hip_runtime.h>
#include <hip/hip_bf16.h>

// ---------------------------------------------------------------------------
// LSTM (H=32, B=64, T=16384, I=O=1) + Linear(32,1) + residual, for gfx1250.
//
// Latency-bound sequential recurrence -> persistent kernel:
//   grid = 4 blocks (16 batch rows each), 64 threads = 2 waves per block.
//   wave hf (0/1) owns hidden columns [hf*16, hf*16+16) -> gate tiles
//   {hf, 2+hf, 4+hf, 6+hf} of the 16x128 gate matrix (PyTorch order i,f,g,o).
//
// Per step, per wave:
//   acc_j  = v_wmma_f32_16x16x4_f32 (A=[x_t,1,0,0], B=[W_ih;bias;0;0])  (exact)
//   acc_j += v_wmma_f32_16x16x32_f16(h_hi, W_hh_tile)                    (f16)
//   acc_j += v_wmma_f32_16x16x32_f16(h_lo, W_hh_tile)   (split-h residual)
//   gates -> c,h in registers; h split to (hi,lo) f16, staged via ping-pong
//   LDS buffers (one s_barrier/step); A-fragment reload = 2x ds_load_b128.
//   Linear(32,1) output = one more WMMA pair against a B with only col 0 set.
// Epilogue kernel adds the residual x into d_out.
// ---------------------------------------------------------------------------

typedef __attribute__((ext_vector_type(16))) _Float16 v16h;
typedef __attribute__((ext_vector_type(8)))  _Float16 v8h;
typedef __attribute__((ext_vector_type(8)))  float    v8f;
typedef __attribute__((ext_vector_type(2)))  float    v2f;

union V16 { v16h v; struct { v8h lo, hi; } s; };

__device__ __forceinline__ float fast_sigmoid(float v) {
    // robust for all v: exp(-v) in [0, inf], rcp(inf)=0
    return __builtin_amdgcn_rcpf(1.0f + __expf(-v));
}
__device__ __forceinline__ float fast_tanh(float v) {
    // tanh(v) = 2*sigmoid(2v) - 1 ; robust at both extremes (no inf/inf)
    float s = __builtin_amdgcn_rcpf(1.0f + __expf(-2.0f * v));
    return fmaf(2.0f, s, -1.0f);
}

__global__ __launch_bounds__(64) void lstm_wmma_persistent(
    const float* __restrict__ x,      // (B, T, 1)
    const float* __restrict__ W_ih,   // (128, 1)
    const float* __restrict__ W_hh,   // (128, 32) row-major [g][k]
    const float* __restrict__ b_ih,   // (128)
    const float* __restrict__ b_hh,   // (128)
    const float* __restrict__ W_lin,  // (1, 32)
    const float* __restrict__ b_lin,  // (1)
    float* __restrict__ out,          // (B, T, 1): this kernel writes lin+b_lin
    int T)
{
    __shared__ __align__(16) _Float16 sW[128 * 32];      // W_hh in f16, [g][k]
    __shared__ __align__(16) _Float16 sWlin[16 * 32];    // row 0 = W_lin, rest 0
    __shared__ __align__(16) _Float16 sHhi[2][16 * 32];  // ping-pong h (hi part)
    __shared__ __align__(16) _Float16 sHlo[2][16 * 32];  // ping-pong h (lo part)

    const int tid  = threadIdx.x;
    const int lane = tid & 31;
    const int hf   = tid >> 5;          // wave id = hidden-half index (0/1)
    const int l16  = lane & 15;
    const bool lo_half = (lane < 16);
    const int b0   = blockIdx.x * 16;   // batch-row base for this block

    // ---- one-time setup -------------------------------------------------
    for (int idx = tid; idx < 128 * 32; idx += 64)
        sW[idx] = (_Float16)W_hh[idx];
    for (int idx = tid; idx < 16 * 32; idx += 64) {
        int n = idx >> 5, k = idx & 31;
        sWlin[idx] = (n == 0) ? (_Float16)W_lin[k] : (_Float16)0.0f;
    }
    __syncthreads();

    // A/B 16-bit fragment per-lane K chunks: lanes<16 -> K {0..7,16..23},
    // lanes>=16 -> K {8..15,24..31}.  (B assumed mirror of documented A layout.)
    const int k0 = lo_half ? 0 : 8;

    v16h bw[4];          // W_hh B-fragments for this wave's 4 gate tiles
    v16h blin;           // Linear(32,1) B-fragment (col 0 = W_lin)
    #pragma unroll
    for (int j = 0; j < 4; ++j) {
        int g = (2 * j + hf) * 16 + l16;     // gate row = B column N
        V16 tmp;
        tmp.s.lo = *(const v8h*)&sW[g * 32 + k0];
        tmp.s.hi = *(const v8h*)&sW[g * 32 + k0 + 16];
        bw[j] = tmp.v;
    }
    {
        V16 tmp;
        tmp.s.lo = *(const v8h*)&sWlin[l16 * 32 + k0];
        tmp.s.hi = *(const v8h*)&sWlin[l16 * 32 + k0 + 16];
        blin = tmp.v;
    }

    // x/bias path as exact f32 16x16x4: B rows K0=W_ih, K1=(b_ih+b_hh), K2=K3=0
    v2f bx[4];
    #pragma unroll
    for (int j = 0; j < 4; ++j) {
        int g = (2 * j + hf) * 16 + l16;
        bx[j][0] = lo_half ? W_ih[g] : 0.0f;
        bx[j][1] = lo_half ? (b_ih[g] + b_hh[g]) : 0.0f;
    }
    const float blin_s = b_lin[0];

    v8f zero8;
    #pragma unroll
    for (int q = 0; q < 8; ++q) zero8[q] = 0.0f;

    // ---- recurrent state ------------------------------------------------
    v16h hhi, hlo;                       // A-fragments of h (split f16 pair)
    #pragma unroll
    for (int q = 0; q < 16; ++q) { hhi[q] = (_Float16)0.0f; hlo[q] = (_Float16)0.0f; }
    float c[8];
    #pragma unroll
    for (int r = 0; r < 8; ++r) c[r] = 0.0f;

    const float* xrow = x + (size_t)(b0 + l16) * (size_t)T;
    float4 xv = *(const float4*)xrow;    // x for t = 0..3 (double-buffered)
    int p = 0;                           // ping-pong index
    const int kst = hf * 16 + l16;       // this lane's hidden column for h store
    const int mb  = lo_half ? 0 : 8;     // D-layout batch-row base per lane group

    for (int t4 = 0; t4 < T; t4 += 4) {
        float4 xnext = xv;
        if (t4 + 4 < T) xnext = *(const float4*)(xrow + t4 + 4);
        float xs[4] = {xv.x, xv.y, xv.z, xv.w};

        #pragma unroll
        for (int u = 0; u < 4; ++u) {
            const int t = t4 + u;

            // A-fragment for the 16x16x4 f32 x-path: [x_t, 1, 0, 0]
            v2f ax;
            ax[0] = lo_half ? xs[u] : 0.0f;   // K=0 rows (lanes 0-15), K=2 zeros
            ax[1] = lo_half ? 1.0f  : 0.0f;   // K=1 rows,              K=3 zeros

            v8f acc[4];
            #pragma unroll
            for (int j = 0; j < 4; ++j) {
                acc[j] = __builtin_amdgcn_wmma_f32_16x16x4_f32(
                    false, ax, false, bx[j], (short)0, zero8, false, false);
                acc[j] = __builtin_amdgcn_wmma_f32_16x16x32_f16(
                    false, hhi, false, bw[j], (short)0, acc[j], false, false);
                acc[j] = __builtin_amdgcn_wmma_f32_16x16x32_f16(
                    false, hlo, false, bw[j], (short)0, acc[j], false, false);
            }

            // gates + state update (D layout: VGPR r -> batch row mb+r, col l16)
            #pragma unroll
            for (int r = 0; r < 8; ++r) {
                float iv = fast_sigmoid(acc[0][r]);
                float fv = fast_sigmoid(acc[1][r]);
                float gv = fast_tanh   (acc[2][r]);
                float ov = fast_sigmoid(acc[3][r]);
                float cn = fmaf(fv, c[r], iv * gv);
                c[r] = cn;
                float hv = ov * fast_tanh(cn);
                _Float16 hh = (_Float16)hv;
                _Float16 hl = (_Float16)(hv - (float)hh);  // split residual
                int m = mb + r;
                sHhi[p][m * 32 + kst] = hh;
                sHlo[p][m * 32 + kst] = hl;
            }
            __syncthreads();   // single barrier/step (ping-pong covers WAR)

            // reload full h as A-fragments (2x ds_load_b128 per fragment)
            {
                V16 a, b;
                a.s.lo = *(const v8h*)&sHhi[p][l16 * 32 + k0];
                a.s.hi = *(const v8h*)&sHhi[p][l16 * 32 + k0 + 16];
                hhi = a.v;
                b.s.lo = *(const v8h*)&sHlo[p][l16 * 32 + k0];
                b.s.hi = *(const v8h*)&sHlo[p][l16 * 32 + k0 + 16];
                hlo = b.v;
            }

            // Linear(32,1) via WMMA; result lives in lanes 0 (m=0..7) / 16 (m=8..15)
            if (hf == 0) {     // wave-uniform branch
                v8f lin = __builtin_amdgcn_wmma_f32_16x16x32_f16(
                    false, hhi, false, blin, (short)0, zero8, false, false);
                lin = __builtin_amdgcn_wmma_f32_16x16x32_f16(
                    false, hlo, false, blin, (short)0, lin, false, false);
                if (l16 == 0) {
                    #pragma unroll
                    for (int r = 0; r < 8; ++r)
                        out[(size_t)(b0 + mb + r) * (size_t)T + t] = lin[r] + blin_s;
                }
            }
            p ^= 1;
        }
        xv = xnext;
    }
}

// residual: out += x  (out was fully overwritten by the main kernel)
__global__ void add_residual(const float* __restrict__ x,
                             float* __restrict__ out, int n) {
    int i = blockIdx.x * blockDim.x + threadIdx.x;
    if (i < n) out[i] += x[i];
}

extern "C" void kernel_launch(void* const* d_in, const int* in_sizes, int n_in,
                              void* d_out, int out_size, void* d_ws, size_t ws_size,
                              hipStream_t stream) {
    const float* x     = (const float*)d_in[0];
    const float* W_ih  = (const float*)d_in[1];
    const float* W_hh  = (const float*)d_in[2];
    const float* b_ih  = (const float*)d_in[3];
    const float* b_hh  = (const float*)d_in[4];
    const float* W_lin = (const float*)d_in[5];
    const float* b_lin = (const float*)d_in[6];
    float* out = (float*)d_out;

    const int B  = 64;
    const int BT = in_sizes[0];      // B*T*I, I=1
    const int T  = BT / B;

    lstm_wmma_persistent<<<B / 16, 64, 0, stream>>>(
        x, W_ih, W_hh, b_ih, b_hh, W_lin, b_lin, out, T);
    add_residual<<<(BT + 255) / 256, 256, 0, stream>>>(x, out, BT);
}
